// CrossEncoderGNN_31860067402182
// MI455X (gfx1250) — matrix operand order repaired and checked by reading, not compiled, not run on hardware
//
#include <hip/hip_runtime.h>
#include <hip/hip_bf16.h>
#include <cstdint>
#include <cstddef>

// ---------------------------------------------------------------------------
// CrossEncoderGNN on MI455X (gfx1250)
//   2x GIN layer: agg = scatter_add(h[src] -> dst); h' = (h+agg) @ W + b
//   pool = segment_sum over batch; head = sigmoid(relu(pool@Wc1+b)@Wc2+b)
// GEMMs run on v_wmma_f32_16x16x32_bf16 (f32 accumulate), software-pipelined:
// each wave owns a 32x64 output strip (2 M-tiles x 4 N-tiles) so every
// K-step does 6 fragment loads feeding 8 WMMAs, with next-step fragments
// loaded before the current WMMAs issue (no per-WMMA loadcnt-0 stall).
// Edge gather/scatter is L2-resident (41MB node matrix << 192MB L2) and uses
// hw f32 atomics.
// ---------------------------------------------------------------------------

#define NNODES 20000
#define NEDGES 320000
#define DIM    512
#define NGRAPH 64
#define MTILES (NNODES / 16)   // 1250
#define KSTEPS (DIM / 32)      // 16
#define NTILES (DIM / 16)      // 32

typedef __attribute__((ext_vector_type(16))) __bf16 v16bf;
typedef __attribute__((ext_vector_type(8)))  float  v8f;

// ---------------- zero fill (float4 granularity, exact count) --------------
__global__ __launch_bounds__(256) void k_zero_f4(float4* __restrict__ p, int n4) {
    int t = blockIdx.x * 256 + threadIdx.x;
    if (t < n4) p[t] = make_float4(0.f, 0.f, 0.f, 0.f);
}

// ---------------- edge scatter-add: agg[dst] += h[src] ---------------------
// 128 threads per edge, float4 per thread. Atomics land in L2 (working set
// fits), so this is L2-bw bound as intended.
__global__ __launch_bounds__(256) void k_scatter_add(const float* __restrict__ h,
                                                     const int* __restrict__ src,
                                                     const int* __restrict__ dst,
                                                     float* __restrict__ agg) {
    int t  = blockIdx.x * 256 + threadIdx.x;     // NEDGES*128 threads total
    int e  = t >> 7;
    int d  = (t & 127) << 2;
    int s  = src[e];
    int dd = dst[e];
    const float4 v = *reinterpret_cast<const float4*>(h + (size_t)s * DIM + d);
    float* p = agg + (size_t)dd * DIM + d;
    unsafeAtomicAdd(p + 0, v.x);
    unsafeAtomicAdd(p + 1, v.y);
    unsafeAtomicAdd(p + 2, v.z);
    unsafeAtomicAdd(p + 3, v.w);
}

// ---------------- pack W (f32 row-major KxN) into B-fragment layout --------
// B 16-bit 32x16 fragment: lane l -> col n = l&15, khalf = l>>4;
// vgpr v holds k = s*32 + khalf*8 + (v<4 ? 2v : 16+2(v-4)), +1.
__global__ __launch_bounds__(256) void k_pack_w(const float* __restrict__ W,
                                                unsigned* __restrict__ outu) {
    int t  = blockIdx.x * 256 + threadIdx.x;     // 32*16*32*8 = 131072 threads
    int v  = t & 7;
    int l  = (t >> 3) & 31;
    int s  = (t >> 8) & 15;
    int nt = t >> 12;                            // 0..31
    int n  = nt * 16 + (l & 15);
    int k  = s * 32 + (l >> 4) * 8 + ((v < 4) ? 2 * v : 16 + 2 * (v - 4));
    float f0 = W[(size_t)k * DIM + n];
    float f1 = W[(size_t)(k + 1) * DIM + n];
    union { __bf16 b[2]; unsigned u; } pk;
    pk.b[0] = (__bf16)f0;
    pk.b[1] = (__bf16)f1;
    outu[((size_t)((nt * 16 + s) * 32 + l)) * 8 + v] = pk.u;
}

// ---------------- pack A = (h + agg) into A-fragment layout ----------------
// A 16-bit 16x32 fragment: lane l -> row m = mt*16 + (l&15), khalf = l>>4;
// vgpr v holds k = s*32 + khalf*8 + (v<4 ? 2v : 16+2(v-4)), +1.
__global__ __launch_bounds__(256) void k_pack_a(const float* __restrict__ h,
                                                const float* __restrict__ agg,
                                                unsigned* __restrict__ outu) {
    int t  = blockIdx.x * 256 + threadIdx.x;     // MTILES*4096 = 5,120,000
    int v  = t & 7;
    int l  = (t >> 3) & 31;
    int s  = (t >> 8) & 15;
    int mt = t >> 12;                            // 0..1249
    int m  = mt * 16 + (l & 15);
    int kb = s * 32 + (l >> 4) * 8 + ((v < 4) ? 2 * v : 16 + 2 * (v - 4));
    size_t idx = (size_t)m * DIM + kb;
    float f0 = h[idx] + agg[idx];
    float f1 = h[idx + 1] + agg[idx + 1];
    union { __bf16 b[2]; unsigned u; } pk;
    pk.b[0] = (__bf16)f0;
    pk.b[1] = (__bf16)f1;
    outu[((size_t)((mt * 16 + s) * 32 + l)) * 8 + v] = pk.u;
}

// ---------------- WMMA GEMM: out[M,512] = Apack @ Wpack + bias -------------
// 625 blocks x 8 waves. Wave w owns M rows [mt0*16, mt0*16+32) and N strip
// [w*64, w*64+64): 2 M-tiles x 4 N-tiles = 8 accumulators. Per K-step:
// 2 A frags + 4 B frags (32B/lane each, coalesced) -> 8 WMMAs, with the next
// step's fragments loaded ahead of the current WMMAs (software pipeline).
__global__ __launch_bounds__(256) void k_gemm_wmma(const __bf16* __restrict__ Apack,
                                                   const __bf16* __restrict__ Wpack,
                                                   const float* __restrict__ bias,
                                                   float* __restrict__ out) {
    const int mt0  = blockIdx.x * 2;             // first of 2 m-tiles
    const int lane = threadIdx.x & 31;
    const int wave = threadIdx.x >> 5;           // 0..7
    const int nt0  = wave * 4;

    // fragment f = tile*16 + s lives at v16bf index f*32 + lane
    const v16bf* A0 = reinterpret_cast<const v16bf*>(Apack) + (size_t)(mt0 + 0) * 16 * 32 + lane;
    const v16bf* A1 = reinterpret_cast<const v16bf*>(Apack) + (size_t)(mt0 + 1) * 16 * 32 + lane;
    const v16bf* Bf = reinterpret_cast<const v16bf*>(Wpack) + lane;

    v8f acc[2][4];
#pragma unroll
    for (int mi = 0; mi < 2; ++mi)
#pragma unroll
        for (int c = 0; c < 4; ++c) acc[mi][c] = v8f{};

    // prologue: load K-step 0
    v16bf a0 = A0[0];
    v16bf a1 = A1[0];
    v16bf b[4];
#pragma unroll
    for (int c = 0; c < 4; ++c) b[c] = Bf[((size_t)(nt0 + c) * 16) * 32];

#pragma unroll 3
    for (int s = 0; s < KSTEPS - 1; ++s) {
        // issue next-step loads first; WMMAs below don't depend on them
        v16bf a0n = A0[(size_t)(s + 1) * 32];
        v16bf a1n = A1[(size_t)(s + 1) * 32];
        v16bf bn[4];
#pragma unroll
        for (int c = 0; c < 4; ++c) bn[c] = Bf[((size_t)(nt0 + c) * 16 + s + 1) * 32];

#pragma unroll
        for (int c = 0; c < 4; ++c)
            acc[0][c] = __builtin_amdgcn_wmma_f32_16x16x32_bf16(
                false, a0, false, b[c], (short)0, acc[0][c], false, false);
#pragma unroll
        for (int c = 0; c < 4; ++c)
            acc[1][c] = __builtin_amdgcn_wmma_f32_16x16x32_bf16(
                false, a1, false, b[c], (short)0, acc[1][c], false, false);

        a0 = a0n;
        a1 = a1n;
#pragma unroll
        for (int c = 0; c < 4; ++c) b[c] = bn[c];
    }
    // epilogue: last K-step
#pragma unroll
    for (int c = 0; c < 4; ++c)
        acc[0][c] = __builtin_amdgcn_wmma_f32_16x16x32_bf16(
            false, a0, false, b[c], (short)0, acc[0][c], false, false);
#pragma unroll
    for (int c = 0; c < 4; ++c)
        acc[1][c] = __builtin_amdgcn_wmma_f32_16x16x32_bf16(
            false, a1, false, b[c], (short)0, acc[1][c], false, false);

    // D layout: vgpr r -> m = mt*16 + (lane>>4)*8 + r ; n = ntile*16 + (lane&15)
    const int nc = lane & 15;
#pragma unroll
    for (int mi = 0; mi < 2; ++mi) {
        const int m0 = (mt0 + mi) * 16 + (lane >> 4) * 8;
#pragma unroll
        for (int c = 0; c < 4; ++c) {
            const int n = (nt0 + c) * 16 + nc;
            const float bn = bias[n];
#pragma unroll
            for (int r = 0; r < 8; ++r) {
                out[(size_t)(m0 + r) * DIM + n] = acc[mi][c][r] + bn;
            }
        }
    }
}

// ---------------- graph pooling: pooled[batch[n]] += h[n] ------------------
__global__ __launch_bounds__(256) void k_pool(const float* __restrict__ h,
                                              const int* __restrict__ batch,
                                              float* __restrict__ pooled) {
    int t = blockIdx.x * 256 + threadIdx.x;      // NNODES*128 threads
    int n = t >> 7;
    int d = (t & 127) << 2;
    int g = batch[n];
    const float4 v = *reinterpret_cast<const float4*>(h + (size_t)n * DIM + d);
    float* p = pooled + (size_t)g * DIM + d;
    unsafeAtomicAdd(p + 0, v.x);
    unsafeAtomicAdd(p + 1, v.y);
    unsafeAtomicAdd(p + 2, v.z);
    unsafeAtomicAdd(p + 3, v.w);
}

// ---------------- MLP head: sigmoid(relu(pool@Wc1+b1)@Wc2+b2) --------------
__global__ __launch_bounds__(256) void k_head(const float* __restrict__ pooled,
                                              const float* __restrict__ Wc1,
                                              const float* __restrict__ bc1,
                                              const float* __restrict__ Wc2,
                                              const float* __restrict__ bc2,
                                              float* __restrict__ out) {
    const int g = blockIdx.x;                    // 0..63
    const int j = threadIdx.x;                   // 0..255
    const float* pr = pooled + (size_t)g * DIM;
    float acc = bc1[j];
#pragma unroll 8
    for (int k = 0; k < DIM; ++k) acc = fmaf(pr[k], Wc1[(size_t)k * 256 + j], acc);
    float z = fmaxf(acc, 0.f);
    __shared__ float red[256];
    red[j] = z * Wc2[j];
    __syncthreads();
    for (int s = 128; s > 0; s >>= 1) {
        if (j < s) red[j] += red[j + s];
        __syncthreads();
    }
    if (j == 0) {
        float sc = red[0] + bc2[0];
        out[g] = 1.f / (1.f + __expf(-sc));
    }
}

// ---------------------------------------------------------------------------
extern "C" void kernel_launch(void* const* d_in, const int* in_sizes, int n_in,
                              void* d_out, int out_size, void* d_ws, size_t ws_size,
                              hipStream_t stream) {
    (void)in_sizes; (void)n_in; (void)out_size; (void)ws_size;

    const float* x     = (const float*)d_in[0];
    const int*   eidx  = (const int*)d_in[1];
    const int*   batch = (const int*)d_in[2];
    const float* Wg1   = (const float*)d_in[3];
    const float* bg1   = (const float*)d_in[4];
    const float* Wg2   = (const float*)d_in[5];
    const float* bg2   = (const float*)d_in[6];
    const float* Wc1   = (const float*)d_in[7];
    const float* bc1   = (const float*)d_in[8];
    const float* Wc2   = (const float*)d_in[9];
    const float* bc2   = (const float*)d_in[10];
    const int* src = eidx;
    const int* dst = eidx + NEDGES;
    float* out = (float*)d_out;

    // ---- workspace carve-up (all offsets 256B aligned) ----
    char* ws = (char*)d_ws;
    const size_t NB = (size_t)NNODES * DIM * sizeof(float);      // 40,960,000
    float*   B0     = (float*)(ws);                              // agg
    float*   B1     = (float*)(ws + NB);                         // h1 / h2
    __bf16*  Apack  = (__bf16*)(ws + 2 * NB);                    // 20,480,000 B
    __bf16*  W1p    = (__bf16*)(ws + 2 * NB + NB / 2);           // 524,288 B
    __bf16*  W2p    = (__bf16*)(ws + 2 * NB + NB / 2 + 524288);
    float*   pooled = (float*)(ws + 2 * NB + NB / 2 + 2 * 524288);

    const int n4_nodes  = NNODES * DIM / 4;                      // 2,560,000
    const int zb_nodes  = n4_nodes / 256;                        // 10,000
    const int scat_blk  = NEDGES * 128 / 256;                    // 160,000
    const int packa_blk = MTILES * 16;                           // 20,000
    const int packw_blk = 512;                                   // 131072/256
    const int pool_blk  = NNODES * 128 / 256;                    // 10,000
    const int gemm_blk  = MTILES / 2;                            // 625

    // weight packs (independent of node data)
    k_pack_w<<<packw_blk, 256, 0, stream>>>(Wg1, (unsigned*)W1p);
    k_pack_w<<<packw_blk, 256, 0, stream>>>(Wg2, (unsigned*)W2p);

    // ---- layer 1 ----
    k_zero_f4<<<zb_nodes, 256, 0, stream>>>((float4*)B0, n4_nodes);
    k_scatter_add<<<scat_blk, 256, 0, stream>>>(x, src, dst, B0);
    k_pack_a<<<packa_blk, 256, 0, stream>>>(x, B0, (unsigned*)Apack);
    k_gemm_wmma<<<gemm_blk, 256, 0, stream>>>(Apack, W1p, bg1, B1);   // h1 -> B1

    // ---- layer 2 ----
    k_zero_f4<<<zb_nodes, 256, 0, stream>>>((float4*)B0, n4_nodes);
    k_scatter_add<<<scat_blk, 256, 0, stream>>>(B1, src, dst, B0);
    k_pack_a<<<packa_blk, 256, 0, stream>>>(B1, B0, (unsigned*)Apack);
    k_gemm_wmma<<<gemm_blk, 256, 0, stream>>>(Apack, W2p, bg2, B1);   // h2 -> B1

    // ---- pool + head ----
    k_zero_f4<<<(NGRAPH * DIM / 4 + 255) / 256, 256, 0, stream>>>((float4*)pooled,
                                                                  NGRAPH * DIM / 4);
    k_pool<<<pool_blk, 256, 0, stream>>>(B1, batch, pooled);
    k_head<<<NGRAPH, 256, 0, stream>>>(pooled, Wc1, bc1, Wc2, bc2, out);
}
